// TransformerEncoderLayer_46986942218374
// MI455X (gfx1250) — matrix operand order repaired and checked by reading, not compiled, not run on hardware
//
#include <hip/hip_runtime.h>
#include <hip/hip_bf16.h>
#include <math.h>

// ---------------------------------------------------------------------------
// CDNA5 (gfx1250) transformer encoder layer.
// All matmuls use v_wmma_f32_16x16x32_bf16 (f32 accumulate, bf16 operands).
// wave32; blocks of 128 threads = 4 waves.
// ---------------------------------------------------------------------------

typedef __bf16 bf16;
typedef __attribute__((ext_vector_type(16))) __bf16 v16bf;
typedef __attribute__((ext_vector_type(8)))  float  v8f;

#define WMMA_BF16(a, b, c) \
  __builtin_amdgcn_wmma_f32_16x16x32_bf16(false, (a), false, (b), (short)0, (c), false, false)

__device__ __forceinline__ bf16 f2bf(float f) {
  unsigned u = __float_as_uint(f);
  u += 0x7fffu + ((u >> 16) & 1u);          // round-to-nearest-even
  unsigned short s = (unsigned short)(u >> 16);
  return __builtin_bit_cast(bf16, s);
}

// A-fragment (16x32 bf16): lane l -> m = l&15, half = l>>4.
// element e -> k = (e>>3)*16 + half*8 + (e&7)  => two contiguous 16B LDS loads.
__device__ __forceinline__ v16bf ld_frag_a(const bf16* p, int half) {
  union { v16bf v; uint4 q[2]; } u;
  u.q[0] = *(const uint4*)(p + half * 8);
  u.q[1] = *(const uint4*)(p + 16 + half * 8);
  return u.v;
}
// B-fragment (32x16 bf16): lane l -> n = l&15, half = l>>4.
// element e -> k = (e>>3)*16 + half*8 + (e&7)  => two contiguous 16B LDS loads from row n.
__device__ __forceinline__ v16bf ld_frag_b(const bf16* p, int half) {
  union { v16bf v; uint4 q[2]; } u;
  u.q[0] = *(const uint4*)(p + half * 8);
  u.q[1] = *(const uint4*)(p + 16 + half * 8);
  return u.v;
}

// ---------------------------------------------------------------------------
// GEMM: C[m, coff+n] = A[m,:] . W[n,:] (+ bias[n]) (+ReLU)
// A: [M,K] f32 row-major, W: [N,K] f32 row-major (i.e. computes A @ W^T).
// Block tile 64x64, BK=32, 4 waves each computing a 32x32 quadrant.
// ---------------------------------------------------------------------------
__global__ __launch_bounds__(128) void gemm_kernel(
    const float* __restrict__ A, const float* __restrict__ Wt,
    const float* __restrict__ bias, float* __restrict__ C,
    int K, int ldc, int coff, int relu)
{
  __shared__ __align__(16) bf16 As[64][40];
  __shared__ __align__(16) bf16 Bs[64][40];
  const int tid  = threadIdx.x;
  const int lane = tid & 31;
  const int wv   = tid >> 5;
  const int wr   = (wv >> 1) * 32;   // wave row base within block tile
  const int wc   = (wv & 1) * 32;    // wave col base
  const size_t bm = (size_t)blockIdx.y * 64;
  const size_t bn = (size_t)blockIdx.x * 64;
  const int half = lane >> 4, l15 = lane & 15;
  const int rr = tid >> 3;           // staging row 0..15
  const int cc = (tid & 7) * 4;      // staging col 0..28

  v8f acc[2][2] = {};

  for (int k0 = 0; k0 < K; k0 += 32) {
    for (int it = 0; it < 4; ++it) {
      const int r = rr + it * 16;
      const float4 va = *(const float4*)(A  + (bm + r) * (size_t)K + k0 + cc);
      As[r][cc+0] = f2bf(va.x); As[r][cc+1] = f2bf(va.y);
      As[r][cc+2] = f2bf(va.z); As[r][cc+3] = f2bf(va.w);
      const float4 vb = *(const float4*)(Wt + (bn + r) * (size_t)K + k0 + cc);
      Bs[r][cc+0] = f2bf(vb.x); Bs[r][cc+1] = f2bf(vb.y);
      Bs[r][cc+2] = f2bf(vb.z); Bs[r][cc+3] = f2bf(vb.w);
    }
    __syncthreads();
    v16bf a0 = ld_frag_a(&As[wr + l15][0], half);
    v16bf a1 = ld_frag_a(&As[wr + 16 + l15][0], half);
    v16bf b0 = ld_frag_b(&Bs[wc + l15][0], half);
    v16bf b1 = ld_frag_b(&Bs[wc + 16 + l15][0], half);
    acc[0][0] = WMMA_BF16(a0, b0, acc[0][0]);
    acc[0][1] = WMMA_BF16(a0, b1, acc[0][1]);
    acc[1][0] = WMMA_BF16(a1, b0, acc[1][0]);
    acc[1][1] = WMMA_BF16(a1, b1, acc[1][1]);
    __syncthreads();
  }

  // C/D layout: element r -> m = 8*half + r, n = lane&15
  for (int rt = 0; rt < 2; ++rt)
    for (int ct = 0; ct < 2; ++ct) {
      const int nl = wc + ct * 16 + l15;
      const float bv = bias ? bias[bn + nl] : 0.0f;
      for (int r = 0; r < 8; ++r) {
        const size_t mg = bm + wr + rt * 16 + half * 8 + r;
        float v = acc[rt][ct][r] + bv;
        if (relu) v = fmaxf(v, 0.0f);
        C[mg * (size_t)ldc + coff + bn + nl] = v;
      }
    }
}

// ---------------------------------------------------------------------------
// Flash attention (shared global/local). qkv: [B,S,1536] with q|k|v blocks of
// 512, head h at column h*64. out: [B,S,512]. mode 0: full softmax over S.
// mode 1: windowed (W=128, look-back 1 window, mask 0 <= kpos <= qpos),
// q pre-scaled, scale=1.
// Block: 64 query rows, 4 waves x 16 rows. K/V tiles of 64.
// ---------------------------------------------------------------------------
__global__ __launch_bounds__(128) void attn_kernel(
    const float* __restrict__ qkv, float* __restrict__ out,
    int mode, float scale)
{
  __shared__ __align__(16) bf16 Qs[64][72];
  __shared__ __align__(16) bf16 Ks[64][72];
  __shared__ __align__(16) bf16 Vt[64][72];       // transposed: Vt[d][j]
  __shared__ __align__(16) bf16 Ps[4][16][72];    // per-wave P tile

  const int tid  = threadIdx.x;
  const int lane = tid & 31;
  const int wv   = tid >> 5;
  const int half = lane >> 4;
  const int l15  = lane & 15;
  const int bh   = blockIdx.y;
  const int b    = bh >> 3;   // H = 8
  const int h    = bh & 7;

  int qbase, kstart, ktiles;
  if (mode == 0) { qbase = blockIdx.x * 64; kstart = 0; ktiles = 2048 / 64; }
  else {
    const int win = blockIdx.x >> 1;
    qbase  = win * 128 + (blockIdx.x & 1) * 64;
    kstart = win * 128 - 128;
    ktiles = 4;
  }
  const float* qp = qkv + (size_t)b * 2048 * 1536 + h * 64;
  const float* kp = qp + 512;
  const float* vp = qp + 1024;

  { // stage Q tile (64 x 64)
    const int r = tid >> 1, ch = (tid & 1) * 32;
    const float* s = qp + (size_t)(qbase + r) * 1536 + ch;
    for (int j = 0; j < 8; ++j) {
      float4 v = *(const float4*)(s + j * 4);
      Qs[r][ch+j*4+0] = f2bf(v.x); Qs[r][ch+j*4+1] = f2bf(v.y);
      Qs[r][ch+j*4+2] = f2bf(v.z); Qs[r][ch+j*4+3] = f2bf(v.w);
    }
  }

  float mrow[8], lrow[8];
  v8f o[4] = {};
  for (int r = 0; r < 8; ++r) { mrow[r] = -3.0e38f; lrow[r] = 0.0f; }

  for (int kt = 0; kt < ktiles; ++kt) {
    __syncthreads();                 // protect Ks/Vt from previous iteration
    const int kb = kstart + kt * 64;
    { // stage K rows + V transposed
      const int r = tid >> 1, ch = (tid & 1) * 32;
      int kg = kb + r; if (kg < 0) kg = 0;   // masked anyway
      const float* ks = kp + (size_t)kg * 1536 + ch;
      const float* vs = vp + (size_t)kg * 1536 + ch;
      for (int j = 0; j < 8; ++j) {
        float4 kv = *(const float4*)(ks + j * 4);
        Ks[r][ch+j*4+0] = f2bf(kv.x); Ks[r][ch+j*4+1] = f2bf(kv.y);
        Ks[r][ch+j*4+2] = f2bf(kv.z); Ks[r][ch+j*4+3] = f2bf(kv.w);
        float4 vv = *(const float4*)(vs + j * 4);
        Vt[ch+j*4+0][r] = f2bf(vv.x); Vt[ch+j*4+1][r] = f2bf(vv.y);
        Vt[ch+j*4+2][r] = f2bf(vv.z); Vt[ch+j*4+3][r] = f2bf(vv.w);
      }
    }
    __syncthreads();

    // S = Q K^T (16 rows x 64 cols per wave)
    float sc[4][8];
    for (int ct = 0; ct < 4; ++ct) {
      v8f s = {};
      for (int dk = 0; dk < 2; ++dk) {
        v16bf qf = ld_frag_a(&Qs[wv * 16 + l15][dk * 32], half);
        v16bf kf = ld_frag_b(&Ks[ct * 16 + l15][dk * 32], half);
        s = WMMA_BF16(qf, kf, s);
      }
      for (int r = 0; r < 8; ++r) {
        float v = s[r] * scale;
        if (mode == 1) {
          const int iq = qbase + wv * 16 + half * 8 + r;
          const int jg = kb + ct * 16 + l15;
          if (jg < 0 || jg > iq) v = -3.0e38f;
        }
        sc[ct][r] = v;
      }
    }

    // online softmax: row-wise over 16 lanes of each half-wave
    float rmax[8];
    for (int r = 0; r < 8; ++r) rmax[r] = mrow[r];
    for (int ct = 0; ct < 4; ++ct)
      for (int r = 0; r < 8; ++r) rmax[r] = fmaxf(rmax[r], sc[ct][r]);
    for (int off = 1; off < 16; off <<= 1)
      for (int r = 0; r < 8; ++r)
        rmax[r] = fmaxf(rmax[r], __shfl_xor(rmax[r], off, 32));

    float alpha[8], rsum[8];
    for (int r = 0; r < 8; ++r) {
      alpha[r] = __expf(mrow[r] - rmax[r]);
      mrow[r]  = rmax[r];
      rsum[r]  = 0.0f;
    }
    for (int ct = 0; ct < 4; ++ct)
      for (int r = 0; r < 8; ++r) {
        float p = (sc[ct][r] > -1.0e37f) ? __expf(sc[ct][r] - rmax[r]) : 0.0f;
        sc[ct][r] = p;
        rsum[r] += p;
      }
    for (int off = 1; off < 16; off <<= 1)
      for (int r = 0; r < 8; ++r) rsum[r] += __shfl_xor(rsum[r], off, 32);
    for (int r = 0; r < 8; ++r) lrow[r] = lrow[r] * alpha[r] + rsum[r];
    for (int t = 0; t < 4; ++t)
      for (int r = 0; r < 8; ++r) o[t][r] *= alpha[r];

    // stage P (bf16) for use as WMMA A operand
    for (int ct = 0; ct < 4; ++ct)
      for (int r = 0; r < 8; ++r)
        Ps[wv][half * 8 + r][ct * 16 + l15] = f2bf(sc[ct][r]);
    __syncthreads();

    // O += P V
    for (int dt = 0; dt < 4; ++dt)
      for (int jc = 0; jc < 2; ++jc) {
        v16bf pf = ld_frag_a(&Ps[wv][l15][jc * 32], half);
        v16bf vf = ld_frag_b(&Vt[dt * 16 + l15][jc * 32], half);
        o[dt] = WMMA_BF16(pf, vf, o[dt]);
      }
  }

  float invl[8];
  for (int r = 0; r < 8; ++r) invl[r] = (lrow[r] > 0.0f) ? 1.0f / lrow[r] : 0.0f;
  for (int dt = 0; dt < 4; ++dt)
    for (int r = 0; r < 8; ++r) {
      const int iq = qbase + wv * 16 + half * 8 + r;
      out[(size_t)(b * 2048 + iq) * 512 + h * 64 + dt * 16 + l15] = o[dt][r] * invl[r];
    }
}

// ---------------------------------------------------------------------------
// RoPE on local qkv buffer [B,S,1536]: rotate (q,k) halves, pre-scale q.
// One thread per (b,s,h,d') with d' in [0,32).
// ---------------------------------------------------------------------------
__global__ __launch_bounds__(256) void rope_kernel(float* __restrict__ qkv)
{
  const int idx = blockIdx.x * blockDim.x + threadIdx.x;
  const int dp = idx & 31;
  int t = idx >> 5;
  const int h = t & 7;  t >>= 3;
  const int s = t & 2047;
  const int b = t >> 11;
  const float inv = __powf(10000.0f, -(float)dp * (1.0f / 32.0f));
  const float fr = (float)s * inv;
  float sn, cs;
  __sincosf(fr, &sn, &cs);
  float* base = qkv + (size_t)(b * 2048 + s) * 1536 + h * 64;
  const float q1 = base[dp], q2 = base[dp + 32];
  base[dp]      = (q1 * cs - q2 * sn) * 0.125f;   // scale = DH^-0.5
  base[dp + 32] = (q2 * cs + q1 * sn) * 0.125f;
  float* kb = base + 512;
  const float k1 = kb[dp], k2 = kb[dp + 32];
  kb[dp]      = k1 * cs - k2 * sn;
  kb[dp + 32] = k2 * cs + k1 * sn;
}

// ---------------------------------------------------------------------------
// LayerNorm over D=512, one wave per row, optional fused residual add.
// ---------------------------------------------------------------------------
__global__ __launch_bounds__(256) void ln_kernel(
    const float* __restrict__ x, const float* __restrict__ add,
    const float* __restrict__ g, const float* __restrict__ bta,
    float* __restrict__ out)
{
  const int lane = threadIdx.x & 31;
  const int wave = threadIdx.x >> 5;
  const size_t row = (size_t)blockIdx.x * 8 + wave;
  const float* xr = x + row * 512;
  const float* ar = add ? add + row * 512 : nullptr;

  float vals[16];
  float sum = 0.0f;
  for (int j = 0; j < 4; ++j) {
    const int c = (j * 32 + lane) * 4;
    float4 v = *(const float4*)(xr + c);
    if (ar) {
      const float4 a = *(const float4*)(ar + c);
      v.x += a.x; v.y += a.y; v.z += a.z; v.w += a.w;
    }
    vals[j*4+0] = v.x; vals[j*4+1] = v.y; vals[j*4+2] = v.z; vals[j*4+3] = v.w;
    sum += v.x + v.y + v.z + v.w;
  }
  for (int off = 16; off > 0; off >>= 1) sum += __shfl_xor(sum, off, 32);
  const float mean = sum * (1.0f / 512.0f);
  float sq = 0.0f;
  for (int i = 0; i < 16; ++i) { const float d = vals[i] - mean; sq += d * d; }
  for (int off = 16; off > 0; off >>= 1) sq += __shfl_xor(sq, off, 32);
  const float rstd = rsqrtf(sq * (1.0f / 512.0f) + 1e-5f);

  float* orow = out + row * 512;
  for (int j = 0; j < 4; ++j) {
    const int c = (j * 32 + lane) * 4;
    float4 o;
    o.x = (vals[j*4+0] - mean) * rstd * g[c+0] + bta[c+0];
    o.y = (vals[j*4+1] - mean) * rstd * g[c+1] + bta[c+1];
    o.z = (vals[j*4+2] - mean) * rstd * g[c+2] + bta[c+2];
    o.w = (vals[j*4+3] - mean) * rstd * g[c+3] + bta[c+3];
    *(float4*)(orow + c) = o;
  }
}

// ---------------------------------------------------------------------------
extern "C" void kernel_launch(void* const* d_in, const int* in_sizes, int n_in,
                              void* d_out, int out_size, void* d_ws, size_t ws_size,
                              hipStream_t stream)
{
  (void)in_sizes; (void)n_in; (void)out_size; (void)ws_size;
  const float* src        = (const float*)d_in[0];
  const float* in_proj_w  = (const float*)d_in[1];
  const float* in_proj_b  = (const float*)d_in[2];
  const float* out_proj_w = (const float*)d_in[3];
  const float* out_proj_b = (const float*)d_in[4];
  const float* ln_g       = (const float*)d_in[5];
  const float* ln_b       = (const float*)d_in[6];
  const float* qkv_w      = (const float*)d_in[7];
  const float* to_out_w   = (const float*)d_in[8];
  const float* gl_w       = (const float*)d_in[9];
  const float* gl_b       = (const float*)d_in[10];
  const float* n1_g       = (const float*)d_in[11];
  const float* n1_b       = (const float*)d_in[12];
  const float* lin1_w     = (const float*)d_in[13];
  const float* lin1_b     = (const float*)d_in[14];
  const float* lin2_w     = (const float*)d_in[15];
  const float* lin2_b     = (const float*)d_in[16];
  const float* n2_g       = (const float*)d_in[17];
  const float* n2_b       = (const float*)d_in[18];
  float* out = (float*)d_out;
  float* ws  = (float*)d_ws;

  const size_t M = 4 * 2048;           // 8192 rows

  // workspace layout with region reuse (~192 MiB of f32)
  float* qkv_g = ws;                   // [M,1536]; region sized M*2048 ...
  float* ff1   = ws;                   // ... reused later as [M,2048]
  size_t off = M * 2048;
  float* qkv_l = ws + off;  off += M * 1536;
  float* x_ln  = ws + off;  float* src2 = x_ln;  off += M * 512;
  float* go    = ws + off;  float* ff2  = go;    off += M * 512;
  float* lo    = ws + off;  float* h1   = lo;    off += M * 512;
  float* ccat  = ws + off;  off += M * 1024;

  const dim3 blk(128);

  // qkv = src @ in_proj_w^T + b
  gemm_kernel<<<dim3(1536/64, M/64), blk, 0, stream>>>(src, in_proj_w, in_proj_b, qkv_g, 512, 1536, 0, 0);
  // x = LN(src)
  ln_kernel<<<dim3(M/8), dim3(256), 0, stream>>>(src, nullptr, ln_g, ln_b, x_ln);
  // qkv_l = x @ qkv_w^T
  gemm_kernel<<<dim3(1536/64, M/64), blk, 0, stream>>>(x_ln, qkv_w, nullptr, qkv_l, 512, 1536, 0, 0);
  // RoPE(q,k) + q*=scale
  rope_kernel<<<dim3((4*2048*8*32)/256), dim3(256), 0, stream>>>(qkv_l);
  // global full attention -> go
  attn_kernel<<<dim3(32, 32), blk, 0, stream>>>(qkv_g, go, 0, 0.125f);
  // local windowed attention -> lo
  attn_kernel<<<dim3(32, 32), blk, 0, stream>>>(qkv_l, lo, 1, 1.0f);
  // concat[:,0:512]   = go @ out_proj_w^T + b
  gemm_kernel<<<dim3(512/64, M/64), blk, 0, stream>>>(go, out_proj_w, out_proj_b, ccat, 512, 1024, 0, 0);
  // concat[:,512:1024]= lo @ to_out_w^T
  gemm_kernel<<<dim3(512/64, M/64), blk, 0, stream>>>(lo, to_out_w, nullptr, ccat, 512, 1024, 512, 0);
  // src2 = concat @ gl_w^T + gl_b
  gemm_kernel<<<dim3(512/64, M/64), blk, 0, stream>>>(ccat, gl_w, gl_b, src2, 1024, 512, 0, 0);
  // h1 = LN(src + src2)
  ln_kernel<<<dim3(M/8), dim3(256), 0, stream>>>(src, src2, n1_g, n1_b, h1);
  // ff1 = relu(h1 @ lin1_w^T + b)
  gemm_kernel<<<dim3(2048/64, M/64), blk, 0, stream>>>(h1, lin1_w, lin1_b, ff1, 512, 2048, 0, 1);
  // ff2 = ff1 @ lin2_w^T + b
  gemm_kernel<<<dim3(512/64, M/64), blk, 0, stream>>>(ff1, lin2_w, lin2_b, ff2, 2048, 512, 0, 0);
  // out = LN(h1 + ff2)
  ln_kernel<<<dim3(M/8), dim3(256), 0, stream>>>(h1, ff2, n2_g, n2_b, out);
}
